// BiGRU_45835890983602
// MI455X (gfx1250) — matrix operand order repaired and checked by reading, not compiled, not run on hardware
//
#include <hip/hip_runtime.h>
#include <hip/hip_bf16.h>

// ---------------------------------------------------------------------------
// BiGRU (S=512, B=64, I=256, H=512, L=2, bidirectional) for gfx1250 / MI455X.
//
//  * Input projections gx = x@Wih^T + bih hoisted into one large bf16 WMMA
//    GEMM per layer (both directions via gridDim.z).
//  * Recurrent path: ONE persistent kernel per layer. 32 co-resident blocks
//    (2 dirs x 16 column slices) loop over all 512 timesteps with an atomic
//    generation-counter grid barrier between steps. Each block stages its
//    96x512 bf16 Whh slice into LDS once (96KB < 320KB/WGP) and reuses it for
//    every step: inner loop is ds_load_b128 -> v_wmma_f32_16x16x32_bf16.
//  * h double-buffered in global (fp32 master + bf16 WMMA copy).
// ---------------------------------------------------------------------------

#define Sx  512
#define Bx  64
#define Ix  256
#define Hx  512
#define G3H 1536   // 3*H
#define NBLK 32    // persistent grid size
#define ROWS 96    // 3 gates * 32 columns per block
#define RSTRIDE 520 // LDS row stride in elements (512 + 8 pad -> 4-bank stride)

typedef __attribute__((ext_vector_type(16))) __bf16 v16bf;
typedef __attribute__((ext_vector_type(8)))  __bf16 v8bf;
typedef __attribute__((ext_vector_type(8)))  float  v8f;

union AFrag { v16bf v; v8bf h[2]; };

__device__ __forceinline__ unsigned short f2bf(float f) {
  union { float f; unsigned u; } c; c.f = f;
  unsigned u = c.u;
  return (unsigned short)((u + 0x7fffu + ((u >> 16) & 1u)) >> 16);
}

__device__ __forceinline__ float sigmoidf_(float x) {
  return 1.0f / (1.0f + __expf(-x));
}

// ---------------------------------------------------------------------------
// fp32 -> bf16 conversion (grid-stride)
// ---------------------------------------------------------------------------
__global__ void k_f32_to_bf16(const float* __restrict__ in,
                              unsigned short* __restrict__ out, int n) {
  int i = blockIdx.x * blockDim.x + threadIdx.x;
  int stride = gridDim.x * blockDim.x;
  for (; i < n; i += stride) out[i] = f2bf(in[i]);
}

// Zero hidden-state double buffers + barrier counters
__global__ void k_zero_h(float* __restrict__ hf, unsigned short* __restrict__ hb,
                         int n, unsigned* __restrict__ sync) {
  int i = blockIdx.x * blockDim.x + threadIdx.x;
  int stride = gridDim.x * blockDim.x;
  for (; i < n; i += stride) { hf[i] = 0.0f; hb[i] = 0; }
  if (blockIdx.x == 0 && threadIdx.x < 2) sync[threadIdx.x] = 0u;
}

// ---------------------------------------------------------------------------
// Input projection GEMM:  gx[M=S*B][N=3H] = Xbf[M][K] @ W[N][K]^T + bias[N]
// Each wave: 2x4 grid of 16x16 WMMA tiles. Block = 8 waves -> 64 x 256.
// ---------------------------------------------------------------------------
__global__ __launch_bounds__(256) void k_gx_gemm(
    const unsigned short* __restrict__ X,
    const unsigned short* __restrict__ Wf, const unsigned short* __restrict__ Wb,
    const float* __restrict__ biasf, const float* __restrict__ biasb,
    float* __restrict__ gxf, float* __restrict__ gxb, int K) {
  const int dir = blockIdx.z;
  const unsigned short* W = dir ? Wb : Wf;
  const float* bias = dir ? biasb : biasf;
  float* gx = dir ? gxb : gxf;

  const int wave = threadIdx.x >> 5;
  const int lane = threadIdx.x & 31;
  const int l15  = lane & 15;
  const int lhi  = lane >> 4;

  const int m_base = blockIdx.x * 64  + (wave & 1) * 32;
  const int n_base = blockIdx.y * 256 + (wave >> 1) * 64;

  const v8f vzero = {0.f, 0.f, 0.f, 0.f, 0.f, 0.f, 0.f, 0.f};
  v8f acc[2][4];
#pragma unroll
  for (int mi = 0; mi < 2; ++mi)
#pragma unroll
    for (int ni = 0; ni < 4; ++ni) acc[mi][ni] = vzero;

  for (int k0 = 0; k0 < K; k0 += 32) {
    AFrag a[2];
#pragma unroll
    for (int mi = 0; mi < 2; ++mi) {
      const unsigned short* ap =
          X + (size_t)(m_base + mi * 16 + l15) * K + k0 + lhi * 16;
      a[mi].h[0] = *(const v8bf*)ap;
      a[mi].h[1] = *(const v8bf*)(ap + 8);
    }
    v16bf b[4];
#pragma unroll
    for (int ni = 0; ni < 4; ++ni) {
      const unsigned short* wp =
          W + (size_t)(n_base + ni * 16 + l15) * K + k0 + lhi * 16;
      b[ni] = *(const v16bf*)wp;
    }
#pragma unroll
    for (int mi = 0; mi < 2; ++mi)
#pragma unroll
      for (int ni = 0; ni < 4; ++ni)
        acc[mi][ni] = __builtin_amdgcn_wmma_f32_16x16x32_bf16(
            false, a[mi].v, false, b[ni], (short)0, acc[mi][ni], false, false);
  }

#pragma unroll
  for (int ni = 0; ni < 4; ++ni) {
    const int n = n_base + ni * 16 + l15;
    const float bv = bias[n];
#pragma unroll
    for (int mi = 0; mi < 2; ++mi) {
#pragma unroll
      for (int r = 0; r < 8; ++r) {
        const int m = m_base + mi * 16 + r + 8 * lhi;
        gx[(size_t)m * G3H + n] = acc[mi][ni][r] + bv;
      }
    }
  }
}

// ---------------------------------------------------------------------------
// Device-wide barrier (generation counter). All NBLK blocks are co-resident.
// ---------------------------------------------------------------------------
__device__ __forceinline__ void grid_sync(unsigned* cnt, unsigned* gen) {
  __syncthreads();
  if (threadIdx.x == 0) {
    __threadfence();
    unsigned g = __hip_atomic_load(gen, __ATOMIC_RELAXED, __HIP_MEMORY_SCOPE_AGENT);
    unsigned old = __hip_atomic_fetch_add(cnt, 1u, __ATOMIC_ACQ_REL,
                                          __HIP_MEMORY_SCOPE_AGENT);
    if (old == NBLK - 1) {
      __hip_atomic_store(cnt, 0u, __ATOMIC_RELAXED, __HIP_MEMORY_SCOPE_AGENT);
      __hip_atomic_fetch_add(gen, 1u, __ATOMIC_RELEASE, __HIP_MEMORY_SCOPE_AGENT);
    } else {
      while (__hip_atomic_load(gen, __ATOMIC_ACQUIRE,
                               __HIP_MEMORY_SCOPE_AGENT) == g) {
        __builtin_amdgcn_s_sleep(1);
      }
    }
    __threadfence();
  }
  __syncthreads();
}

// ---------------------------------------------------------------------------
// Persistent recurrent kernel: all 512 timesteps of one layer, both dirs.
// Grid: 32 blocks = 2 dirs x 16 column-slices (32 hidden cols each).
// LDS: this block's Whh slice, rows {g*H + n0 + jj}, staged once, reused 512x.
// Wave w -> M-tile (w&3), N-tile (w>>2); R/Z/N accumulators per wave.
// ---------------------------------------------------------------------------
__global__ __launch_bounds__(256) void k_gru_persist(
    const float* __restrict__ gxf,            // [S][B][3H]
    const float* __restrict__ gxb,
    const unsigned short* __restrict__ Wf,    // [3H][H] bf16
    const unsigned short* __restrict__ Wb,
    const float* __restrict__ bhf,            // [3H]
    const float* __restrict__ bhb,
    float* h32,                               // [parity][2][B][H] fp32
    unsigned short* hbf,                      // [parity][2][B][H] bf16
    float* __restrict__ out,                  // [S][B][2H]
    unsigned* __restrict__ sync) {            // [cnt, gen]
  extern __shared__ unsigned short lds[];     // ROWS * RSTRIDE bf16

  const int dir = blockIdx.x >> 4;
  const int n0  = (blockIdx.x & 15) * 32;

  const float* gx = dir ? gxb : gxf;
  const unsigned short* W = dir ? Wb : Wf;
  const float* bh = dir ? bhb : bhf;

  const int wave = threadIdx.x >> 5;
  const int lane = threadIdx.x & 31;
  const int l15  = lane & 15;
  const int lhi  = lane >> 4;
  const int mt   = (wave & 3) * 16;          // batch-row tile base
  const int jloc = (wave >> 2) * 16;         // local column tile base (0|16)
  const int j    = n0 + jloc + l15;          // this lane's hidden column

  // ---- stage Whh slice into LDS: 96 rows x 512 elems, 16B chunks ----
  for (int idx = threadIdx.x; idx < ROWS * (Hx / 8); idx += 256) {
    const int row = idx >> 6;                // 0..95
    const int c8  = idx & 63;                // 8-elem chunk
    const int g   = row >> 5;                // gate 0..2
    const int jj  = row & 31;
    *(v8bf*)(lds + row * RSTRIDE + c8 * 8) =
        *(const v8bf*)(W + (size_t)(g * Hx + n0 + jj) * Hx + c8 * 8);
  }
  __syncthreads();

  const float br = bh[j];
  const float bz = bh[Hx + j];
  const float bn = bh[2 * Hx + j];

  const size_t PST = (size_t)2 * Bx * Hx;    // parity stride
  const size_t dof = (size_t)dir * Bx * Hx;

  for (int t = 0; t < Sx; ++t) {
    const int s = dir ? (Sx - 1 - t) : t;
    const float* hp = h32 + (size_t)(t & 1) * PST + dof;
    float* hn       = h32 + (size_t)((t + 1) & 1) * PST + dof;
    const unsigned short* hpb = hbf + (size_t)(t & 1) * PST + dof;
    unsigned short* hnb       = hbf + (size_t)((t + 1) & 1) * PST + dof;

    const size_t gxRow = (size_t)s * Bx * G3H;
    __builtin_prefetch(gx + gxRow + (size_t)(mt + l15) * G3H + n0 + jloc, 0, 0);

    const v8f vzero = {0.f, 0.f, 0.f, 0.f, 0.f, 0.f, 0.f, 0.f};
    v8f aR = vzero, aZ = vzero, aN = vzero;

#pragma unroll 4
    for (int k0 = 0; k0 < Hx; k0 += 32) {
      AFrag a;
      const unsigned short* ap = hpb + (size_t)(mt + l15) * Hx + k0 + lhi * 16;
      a.h[0] = *(const v8bf*)ap;
      a.h[1] = *(const v8bf*)(ap + 8);

      const int koff = k0 + lhi * 16;
      AFrag bR, bZ, bN;
      const unsigned short* l0 = lds + (jloc + l15) * RSTRIDE + koff;
      bR.h[0] = *(const v8bf*)l0;
      bR.h[1] = *(const v8bf*)(l0 + 8);
      const unsigned short* l1 = l0 + 32 * RSTRIDE;
      bZ.h[0] = *(const v8bf*)l1;
      bZ.h[1] = *(const v8bf*)(l1 + 8);
      const unsigned short* l2 = l0 + 64 * RSTRIDE;
      bN.h[0] = *(const v8bf*)l2;
      bN.h[1] = *(const v8bf*)(l2 + 8);

      aR = __builtin_amdgcn_wmma_f32_16x16x32_bf16(false, a.v, false, bR.v,
                                                   (short)0, aR, false, false);
      aZ = __builtin_amdgcn_wmma_f32_16x16x32_bf16(false, a.v, false, bZ.v,
                                                   (short)0, aZ, false, false);
      aN = __builtin_amdgcn_wmma_f32_16x16x32_bf16(false, a.v, false, bN.v,
                                                   (short)0, aN, false, false);
    }

#pragma unroll
    for (int r = 0; r < 8; ++r) {
      const int m = mt + r + 8 * lhi;
      const size_t gbase = gxRow + (size_t)m * G3H + j;
      const float rg = sigmoidf_(gx[gbase] + aR[r] + br);
      const float zg = sigmoidf_(gx[gbase + Hx] + aZ[r] + bz);
      const float ng = tanhf(gx[gbase + 2 * Hx] + rg * (aN[r] + bn));
      const float hprev = hp[(size_t)m * Hx + j];
      const float hnew = (1.0f - zg) * ng + zg * hprev;
      hn[(size_t)m * Hx + j]  = hnew;
      hnb[(size_t)m * Hx + j] = f2bf(hnew);
      out[(size_t)s * Bx * 2 * Hx + (size_t)m * 2 * Hx + dir * Hx + j] = hnew;
    }

    grid_sync(sync, sync + 1);
  }
}

// ---------------------------------------------------------------------------
// Host side
// ---------------------------------------------------------------------------
extern "C" void kernel_launch(void* const* d_in, const int* in_sizes, int n_in,
                              void* d_out, int out_size, void* d_ws, size_t ws_size,
                              hipStream_t stream) {
  (void)in_sizes; (void)n_in; (void)out_size; (void)ws_size;

  const float* input   = (const float*)d_in[0];
  const float* Wih_f0  = (const float*)d_in[1];
  const float* Whh_f0  = (const float*)d_in[2];
  const float* bih_f0  = (const float*)d_in[3];
  const float* bhh_f0  = (const float*)d_in[4];
  const float* Wih_b0  = (const float*)d_in[5];
  const float* Whh_b0  = (const float*)d_in[6];
  const float* bih_b0  = (const float*)d_in[7];
  const float* bhh_b0  = (const float*)d_in[8];
  const float* Wih_f1  = (const float*)d_in[9];
  const float* Whh_f1  = (const float*)d_in[10];
  const float* bih_f1  = (const float*)d_in[11];
  const float* bhh_f1  = (const float*)d_in[12];
  const float* Wih_b1  = (const float*)d_in[13];
  const float* Whh_b1  = (const float*)d_in[14];
  const float* bih_b1  = (const float*)d_in[15];
  const float* bhh_b1  = (const float*)d_in[16];
  float* out = (float*)d_out;

  char* ws = (char*)d_ws;
  size_t off = 0;
  auto carve = [&](size_t bytes) -> char* {
    char* p = ws + off;
    off += (bytes + 255) & ~(size_t)255;
    return p;
  };
  const size_t SB = (size_t)Sx * Bx;

  unsigned short* xbf = (unsigned short*)carve(SB * 2 * Hx * 2);
  unsigned short* wihf0 = (unsigned short*)carve((size_t)G3H * Ix * 2);
  unsigned short* whhf0 = (unsigned short*)carve((size_t)G3H * Hx * 2);
  unsigned short* wihb0 = (unsigned short*)carve((size_t)G3H * Ix * 2);
  unsigned short* whhb0 = (unsigned short*)carve((size_t)G3H * Hx * 2);
  unsigned short* wihf1 = (unsigned short*)carve((size_t)G3H * 2 * Hx * 2);
  unsigned short* whhf1 = (unsigned short*)carve((size_t)G3H * Hx * 2);
  unsigned short* wihb1 = (unsigned short*)carve((size_t)G3H * 2 * Hx * 2);
  unsigned short* whhb1 = (unsigned short*)carve((size_t)G3H * Hx * 2);
  float* gxf  = (float*)carve(SB * G3H * 4);
  float* gxb  = (float*)carve(SB * G3H * 4);
  float* out0 = (float*)carve(SB * 2 * Hx * 4);
  float* hf32 = (float*)carve((size_t)2 * 2 * Bx * Hx * 4);   // [parity][dir][B][H]
  unsigned short* hbf = (unsigned short*)carve((size_t)2 * 2 * Bx * Hx * 2);
  unsigned* syncp = (unsigned*)carve(256);                    // [cnt, gen]

  auto cvt = [&](const float* src, unsigned short* dst, int n) {
    int blocks = (n + 255) / 256; if (blocks > 2048) blocks = 2048;
    k_f32_to_bf16<<<blocks, 256, 0, stream>>>(src, dst, n);
  };

  cvt(input,  xbf,   (int)(SB * Ix));
  cvt(Wih_f0, wihf0, G3H * Ix);
  cvt(Whh_f0, whhf0, G3H * Hx);
  cvt(Wih_b0, wihb0, G3H * Ix);
  cvt(Whh_b0, whhb0, G3H * Hx);
  cvt(Wih_f1, wihf1, G3H * 2 * Hx);
  cvt(Whh_f1, whhf1, G3H * Hx);
  cvt(Wih_b1, wihb1, G3H * 2 * Hx);
  cvt(Whh_b1, whhb1, G3H * Hx);

  const dim3 gxGrid((unsigned)(SB / 64), G3H / 256, 2);
  const int hzN = (int)(2 * 2 * Bx * Hx);
  const size_t ldsBytes = (size_t)ROWS * RSTRIDE * 2;  // 99840 B

  // ================= layer 0 =================
  k_gx_gemm<<<gxGrid, 256, 0, stream>>>(xbf, wihf0, wihb0, bih_f0, bih_b0,
                                        gxf, gxb, Ix);
  k_zero_h<<<512, 256, 0, stream>>>(hf32, hbf, hzN, syncp);
  k_gru_persist<<<NBLK, 256, ldsBytes, stream>>>(
      gxf, gxb, whhf0, whhb0, bhh_f0, bhh_b0, hf32, hbf, out0, syncp);

  // ================= layer 1 =================
  cvt(out0, xbf, (int)(SB * 2 * Hx));
  k_gx_gemm<<<gxGrid, 256, 0, stream>>>(xbf, wihf1, wihb1, bih_f1, bih_b1,
                                        gxf, gxb, 2 * Hx);
  k_zero_h<<<512, 256, 0, stream>>>(hf32, hbf, hzN, syncp);
  k_gru_persist<<<NBLK, 256, ldsBytes, stream>>>(
      gxf, gxb, whhf1, whhb1, bhh_f1, bhh_b1, hf32, hbf, out, syncp);
}
